// SkipDropGCN_15616501088588
// MI455X (gfx1250) — matrix operand order repaired and checked by reading, not compiled, not run on hardware
//
#include <hip/hip_runtime.h>
#include <hip/hip_bf16.h>

typedef _Float16 v16h __attribute__((ext_vector_type(16)));
typedef _Float16 v8h  __attribute__((ext_vector_type(8)));
typedef float    v8f  __attribute__((ext_vector_type(8)));
typedef float    v2f  __attribute__((ext_vector_type(2)));

#define GCN_N  50000
#define GCN_E  800000
#define GCN_K  128

// ---------------------------------------------------------------------------
// WMMA GEMM: out[M x OutReal] = act[M x 128] @ W[OutReal x 128]^T * scale + bias
// One wave per 16-row tile; NT 16-col tiles (NT=8 -> 128 cols, NT=3 -> 48 pad 40).
// W is staged into LDS as f16 (row stride 136 halfs = 272B = 16*17 -> 16B aligned,
// bank-stride 68 % 64 = 4 -> conflict-free across a 16-lane group).
// ---------------------------------------------------------------------------
template <int NT>
__global__ __launch_bounds__(256) void gemm_wmma_k(
    const float* __restrict__ act, const float* __restrict__ W,
    float* __restrict__ out, const float* __restrict__ scaleptr,
    const float* __restrict__ bias, int Mtiles, int OutReal) {
  constexpr int K    = GCN_K;
  constexpr int ROWS = NT * 16;
  constexpr int LSTR = 136;  // halfs per LDS row (128 + pad)
  __shared__ alignas(16) _Float16 sW[ROWS * LSTR];

  const int tid = threadIdx.x;
  // cooperative stage: W (OutReal x 128) -> f16 LDS, zero-fill padded rows
  for (int idx = tid; idx < ROWS * 64; idx += 256) {
    int rrow = idx >> 6;
    int pc   = (idx & 63) * 2;
    float2 wv = make_float2(0.f, 0.f);
    if (rrow < OutReal) wv = *(const float2*)(W + rrow * K + pc);
    sW[rrow * LSTR + pc]     = (_Float16)wv.x;
    sW[rrow * LSTR + pc + 1] = (_Float16)wv.y;
  }
  __syncthreads();

  const int wave = tid >> 5;
  const int lane = tid & 31;
  const int r    = lane & 15;   // row-in-tile / col-in-tile selector
  const int h    = lane >> 4;   // half-wave selector
  const int tile = blockIdx.x * 8 + wave;
  if (tile >= Mtiles) return;   // uniform per wave -> EXEC stays all-ones

  v8f acc[NT];
#pragma unroll
  for (int nt = 0; nt < NT; ++nt)
#pragma unroll
    for (int j = 0; j < 8; ++j) acc[nt][j] = 0.0f;

  const float* arow = act + (size_t)(tile * 16 + r) * K;

#pragma unroll
  for (int ks = 0; ks < 4; ++ks) {  // K = 4 * 32
    // A fragment: 16x32 f16, contiguous k order (k = h*16 + element)
    //   lane r,h ; VGPR v holds K = ks*32 + h*16 + v*2 pair (matches B)
    v16h a;
#pragma unroll
    for (int v = 0; v < 8; ++v) {
      int k = ks * 32 + h * 16 + (v << 1);
      v2f a2 = *(const v2f*)(arow + k);
      a[2 * v]     = (_Float16)a2.x;
      a[2 * v + 1] = (_Float16)a2.y;
    }
#pragma unroll
    for (int nt = 0; nt < NT; ++nt) {
      // B fragment: lane r holds column n = nt*16+r ; K = ks*32 + h*16 .. +15
      // -> 16 consecutive f16 (32B) per lane, 16B aligned
      const _Float16* bp = &sW[(nt * 16 + r) * LSTR + ks * 32 + h * 16];
      v8h blo = *(const v8h*)bp;
      v8h bhi = *(const v8h*)(bp + 8);
      v16h b;
#pragma unroll
      for (int i = 0; i < 8; ++i) { b[i] = blo[i]; b[8 + i] = bhi[i]; }
      acc[nt] = __builtin_amdgcn_wmma_f32_16x16x32_f16(
          false, a, false, b, (short)0, acc[nt], false, false);
    }
  }

  const float s = scaleptr ? scaleptr[0] : 1.0f;
#pragma unroll
  for (int nt = 0; nt < NT; ++nt) {
#pragma unroll
    for (int j = 0; j < 8; ++j) {
      int row = tile * 16 + j + 8 * h;  // C/D layout: VGPR j, half h -> M
      int col = nt * 16 + r;            // N = lane % 16
      if (col < OutReal) {
        float v = acc[nt][j] * s;
        if (bias) v += bias[col];
        out[(size_t)row * OutReal + col] = v;
      }
    }
  }
}

// ---------------------------------------------------------------------------
// Frobenius norm: scale = 1/||W||_F over n elements (single block)
// ---------------------------------------------------------------------------
__global__ __launch_bounds__(256) void frob_k(const float* __restrict__ W, int n,
                                              float* __restrict__ scale) {
  __shared__ float red[256];
  float s = 0.f;
  for (int i = threadIdx.x; i < n; i += 256) { float w = W[i]; s += w * w; }
  red[threadIdx.x] = s;
  __syncthreads();
  for (int off = 128; off > 0; off >>= 1) {
    if (threadIdx.x < off) red[threadIdx.x] += red[threadIdx.x + off];
    __syncthreads();
  }
  if (threadIdx.x == 0) scale[0] = rsqrtf(red[0]);
}

// ---------------------------------------------------------------------------
// degree / dinv
// ---------------------------------------------------------------------------
__global__ __launch_bounds__(256) void deg_zero_k(float* __restrict__ deg, int n) {
  int i = blockIdx.x * 256 + threadIdx.x;
  if (i < n) deg[i] = 0.f;
}
__global__ __launch_bounds__(256) void deg_edge_k(const long long* __restrict__ A,
                                                  float* __restrict__ deg, int E) {
  int e = blockIdx.x * 256 + threadIdx.x;
  if (e < E) {
    int d = (int)A[E + e];
    __hip_atomic_fetch_add(&deg[d], 1.0f, __ATOMIC_RELAXED, __HIP_MEMORY_SCOPE_AGENT);
  }
}
__global__ __launch_bounds__(256) void dinv_k(float* __restrict__ deg, int n) {
  int i = blockIdx.x * 256 + threadIdx.x;
  if (i < n) deg[i] = rsqrtf(deg[i] + 1.0f);
}

// ---------------------------------------------------------------------------
// self-loop + bias init:  AGG[i,f] = b[f] + dinv[i]^2 * H[i,f]
// ---------------------------------------------------------------------------
__global__ __launch_bounds__(256) void init_agg_k(const float* __restrict__ H,
                                                  const float* __restrict__ dinv,
                                                  const float* __restrict__ b,
                                                  float* __restrict__ AGG, int total) {
  int idx = blockIdx.x * 256 + threadIdx.x;
  if (idx < total) {
    int i = idx >> 7, f = idx & 127;
    float d = dinv[i];
    AGG[idx] = b[f] + d * d * H[idx];
  }
}

// ---------------------------------------------------------------------------
// edge scatter: one wave per edge, lane = 4 contiguous features (float4 gather)
// AGG[dst] += dinv[src]*dinv[dst] * H[src]
// ---------------------------------------------------------------------------
__global__ __launch_bounds__(256) void edge_agg_k(const long long* __restrict__ A,
                                                  const float* __restrict__ H,
                                                  const float* __restrict__ dinv,
                                                  float* __restrict__ AGG, int E) {
  int t = blockIdx.x * 256 + threadIdx.x;
  int e = t >> 5;
  int lane = t & 31;
  if (e >= E) return;
  int s = (int)A[e];
  int d = (int)A[E + e];
  float w = dinv[s] * dinv[d];
  const float4* H4 = (const float4*)H;
  float4 hv = H4[(size_t)s * 32 + lane];
  float* base = AGG + (size_t)d * 128 + lane * 4;
  __hip_atomic_fetch_add(base + 0, w * hv.x, __ATOMIC_RELAXED, __HIP_MEMORY_SCOPE_AGENT);
  __hip_atomic_fetch_add(base + 1, w * hv.y, __ATOMIC_RELAXED, __HIP_MEMORY_SCOPE_AGENT);
  __hip_atomic_fetch_add(base + 2, w * hv.z, __ATOMIC_RELAXED, __HIP_MEMORY_SCOPE_AGENT);
  __hip_atomic_fetch_add(base + 3, w * hv.w, __ATOMIC_RELAXED, __HIP_MEMORY_SCOPE_AGENT);
}

// ---------------------------------------------------------------------------
// layer0: ACT = AGG ; layers>=1: ACT += relu(AGG)   (skip == h after each layer)
// ---------------------------------------------------------------------------
__global__ __launch_bounds__(256) void finalize_k(const float* __restrict__ AGG,
                                                  float* __restrict__ ACT, int total,
                                                  int first) {
  int idx = blockIdx.x * 256 + threadIdx.x;
  if (idx < total) {
    float v = AGG[idx];
    if (first) ACT[idx] = v;
    else       ACT[idx] += (v > 0.f ? v : 0.f);
  }
}

// ---------------------------------------------------------------------------
extern "C" void kernel_launch(void* const* d_in, const int* in_sizes, int n_in,
                              void* d_out, int out_size, void* d_ws, size_t ws_size,
                              hipStream_t stream) {
  const float*     X  = (const float*)d_in[0];
  const long long* A  = (const long long*)d_in[1];
  const float* Wl[4]  = {(const float*)d_in[2], (const float*)d_in[4],
                         (const float*)d_in[6], (const float*)d_in[8]};
  const float* bl[4]  = {(const float*)d_in[3], (const float*)d_in[5],
                         (const float*)d_in[7], (const float*)d_in[9]};
  const float* Wf     = (const float*)d_in[10];
  const float* bf     = (const float*)d_in[11];
  float*       out    = (float*)d_out;

  const int N = GCN_N, E = GCN_E, K = GCN_K;
  float* ws    = (float*)d_ws;
  float* deg   = ws;                       // N floats; becomes dinv in place
  float* scale = ws + N;                   // 1 float
  float* H     = ws + N + 64;              // N*K
  float* AGG   = H + (size_t)N * K;        // N*K
  float* ACT   = AGG + (size_t)N * K;      // N*K  (h == skip)

  const int nb = (N + 255) / 256;
  deg_zero_k<<<nb, 256, 0, stream>>>(deg, N);
  deg_edge_k<<<(E + 255) / 256, 256, 0, stream>>>(A, deg, E);
  dinv_k<<<nb, 256, 0, stream>>>(deg, N);

  const int Mtiles  = N / 16;              // 3125 (exact)
  const int gblocks = (Mtiles + 7) / 8;    // 391
  const int total   = N * K;
  const int tb      = (total + 255) / 256;
  const int eb      = (E * 32 + 255) / 256;  // wave per edge

  for (int l = 0; l < 4; ++l) {
    frob_k<<<1, 256, 0, stream>>>(Wl[l], K * K, scale);
    gemm_wmma_k<8><<<gblocks, 256, 0, stream>>>(l == 0 ? X : ACT, Wl[l], H,
                                                scale, nullptr, Mtiles, 128);
    init_agg_k<<<tb, 256, 0, stream>>>(H, deg, bl[l], AGG, total);
    edge_agg_k<<<eb, 256, 0, stream>>>(A, H, deg, AGG, E);
    finalize_k<<<tb, 256, 0, stream>>>(AGG, ACT, total, l == 0 ? 1 : 0);
  }

  // classifier: 128 -> 40 (padded to 48, masked stores), bias fused, no norm
  gemm_wmma_k<3><<<gblocks, 256, 0, stream>>>(ACT, Wf, out, nullptr, bf,
                                              Mtiles, 40);
}
